// SGCN_60730837565907
// MI455X (gfx1250) — matrix operand order repaired and checked by reading, not compiled, not run on hardware
//
#include <hip/hip_runtime.h>
#include <hip/hip_bf16.h>

// ---- problem constants (match reference) ----
#define N_NODES 100000
#define N_EDGES 3200000
#define N_F     128
#define HIDDEN  128
#define N_PRED  8
#define K_HOPS  3
#define N_GRAPHS 64

typedef _Float16 v8h  __attribute__((ext_vector_type(8)));
typedef _Float16 v16h __attribute__((ext_vector_type(16)));
typedef float    v8f  __attribute__((ext_vector_type(8)));

// deg[i] = 1.0 (self-loop weight)
__global__ void k_init_deg(float* __restrict__ deg, int n) {
    int i = blockIdx.x * blockDim.x + threadIdx.x;
    if (i < n) deg[i] = 1.0f;
}

// deg[col[e]] += w[e]
__global__ void k_deg_scatter(const int* __restrict__ col, const float* __restrict__ w,
                              float* __restrict__ deg, int e) {
    int i = blockIdx.x * blockDim.x + threadIdx.x;
    if (i < e) atomicAdd(&deg[col[i]], w[i]);
}

// deg -> dinv (in place)
__global__ void k_dinv(float* __restrict__ deg, int n) {
    int i = blockIdx.x * blockDim.x + threadIdx.x;
    if (i < n) { float d = deg[i]; deg[i] = (d > 0.0f) ? rsqrtf(d) : 0.0f; }
}

// norm[e] = dinv[row]*w*dinv[col]   (cached once, reused all K hops)
__global__ void k_norm(const int* __restrict__ row, const int* __restrict__ col,
                       const float* __restrict__ w, const float* __restrict__ dinv,
                       float* __restrict__ norm, int e) {
    int i = blockIdx.x * blockDim.x + threadIdx.x;
    if (i < e) norm[i] = dinv[row[i]] * w[i] * dinv[col[i]];
}

// dst[i][:] = dinv[i]^2 * src[i][:]   (self-loop term; also zero-initializes dst)
__global__ void k_selfloop_init(const float* __restrict__ src, const float* __restrict__ dinv,
                                float* __restrict__ dst, int n32) {
    int gid = blockIdx.x * blockDim.x + threadIdx.x;
    if (gid >= n32) return;
    int i = gid >> 5;          // node
    int q = gid & 31;          // float4 chunk (128 feats / 4)
    float di = dinv[i];
    float s = di * di;
    const float4 v = reinterpret_cast<const float4*>(src)[(size_t)i * 32 + q];
    float4 o; o.x = s * v.x; o.y = s * v.y; o.z = s * v.z; o.w = s * v.w;
    reinterpret_cast<float4*>(dst)[(size_t)i * 32 + q] = o;
}

// dst[col][:] += norm[e] * src[row][:]
__global__ void k_edge_scatter(const int* __restrict__ row, const int* __restrict__ col,
                               const float* __restrict__ norm, const float* __restrict__ src,
                               float* __restrict__ dst, long long e32) {
    long long gid = (long long)blockIdx.x * blockDim.x + threadIdx.x;
    if (gid >= e32) return;
    int e = (int)(gid >> 5);
    int q = (int)(gid & 31);
    int r = row[e], c = col[e];
    float nm = norm[e];
    const float4 v = reinterpret_cast<const float4*>(src)[(size_t)r * 32 + q];
    float* base = dst + (size_t)c * 128 + q * 4;
    atomicAdd(base + 0, nm * v.x);
    atomicAdd(base + 1, nm * v.y);
    atomicAdd(base + 2, nm * v.z);
    atomicAdd(base + 3, nm * v.w);
}

__global__ void k_f32_to_f16(const float* __restrict__ src, _Float16* __restrict__ dst, int n) {
    int i = blockIdx.x * blockDim.x + threadIdx.x;
    if (i < n) dst[i] = (_Float16)src[i];
}

__global__ void k_zero(float* __restrict__ p, int n) {
    int i = blockIdx.x * blockDim.x + threadIdx.x;
    if (i < n) p[i] = 0.0f;
}

// out[N x 128] = A[N x 128](f16) @ W[128(out) x 128(in)]^T (f16) + bias, f32 accumulate via WMMA.
// 1 wave -> one 16x16 tile. 8 waves/block share a 16-row A slab; wave w does output-feature tile w.
__global__ __launch_bounds__(256) void k_wmma_gemm(const _Float16* __restrict__ A,
                                                   const _Float16* __restrict__ W,
                                                   const float* __restrict__ bias,
                                                   float* __restrict__ out) {
    const int lane    = threadIdx.x & 31;
    const int ntile   = threadIdx.x >> 5;   // 0..7 (HIDDEN/16)
    const int mtile   = blockIdx.x;         // 0..6249
    const int hi      = lane >> 4;          // 0: lanes 0-15, 1: lanes 16-31
    const int lm      = lane & 15;

    const int m = mtile * 16 + lm;          // A row this lane feeds (ISA: M = lane%16)
    const int n = ntile * 16 + lm;          // B column / W row (ISA: N = lane%16)

    v8f c = {};
    #pragma unroll
    for (int kt = 0; kt < 4; ++kt) {
        const int k0 = kt * 32;
        // A 16x32 f16, contiguous k order chosen here: lanes0-15 hold K=0..15; lanes16-31 hold K=16..31
        const _Float16* ap = A + (size_t)m * 128 + k0 + hi * 16;
        v8h alo = *reinterpret_cast<const v8h*>(ap);
        v8h ahi = *reinterpret_cast<const v8h*>(ap + 8);
        v16h a;
        #pragma unroll
        for (int t = 0; t < 8; ++t) { a[t] = alo[t]; a[8 + t] = ahi[t]; }
        // B 32x16 f16, same k order as A: lanes0-15 hold K=0..15 of column n; lanes16-31 hold K=16..31.
        // B[k][n] = W[n][k]  -> 16 contiguous f16 from W row n.
        const _Float16* bp = W + (size_t)n * 128 + k0 + hi * 16;
        v16h b = *reinterpret_cast<const v16h*>(bp);
        c = __builtin_amdgcn_wmma_f32_16x16x32_f16(false, a, false, b, (short)0, c, false, false);
    }
    // C/D layout: VGPR j holds row M = (hi?8:0)+j, column N = lane%16
    const float bv = bias[n];
    const int mbase = mtile * 16 + hi * 8;
    #pragma unroll
    for (int j = 0; j < 8; ++j)
        out[(size_t)(mbase + j) * 128 + ntile * 16 + lm] = c[j] + bv;
}

// pooled sums + counts
__global__ void k_pool(const float* __restrict__ hlin, const int* __restrict__ batch,
                       float* __restrict__ sums, float* __restrict__ cnt, int n128) {
    int gid = blockIdx.x * blockDim.x + threadIdx.x;
    if (gid >= n128) return;
    int i = gid >> 7;
    int f = gid & 127;
    int g = batch[i];
    atomicAdd(&sums[(size_t)g * 128 + f], hlin[gid]);
    if (f == 0) atomicAdd(&cnt[g], 1.0f);
}

// out[g][p] = (sums[g]/max(cnt,1)) . W_lin[p] + b_lin[p]
__global__ void k_final(const float* __restrict__ sums, const float* __restrict__ cnt,
                        const float* __restrict__ Wl, const float* __restrict__ bl,
                        float* __restrict__ out) {
    int tid = blockIdx.x * blockDim.x + threadIdx.x;
    if (tid >= N_GRAPHS * N_PRED) return;
    int g = tid >> 3;
    int p = tid & 7;
    float c = cnt[g];
    float inv = 1.0f / (c > 1.0f ? c : 1.0f);
    float acc = bl[p];
    #pragma unroll 8
    for (int k = 0; k < HIDDEN; ++k)
        acc += (sums[(size_t)g * 128 + k] * inv) * Wl[(size_t)p * 128 + k];
    out[tid] = acc;
}

static inline int cdiv(long long a, long long b) { return (int)((a + b - 1) / b); }

extern "C" void kernel_launch(void* const* d_in, const int* in_sizes, int n_in,
                              void* d_out, int out_size, void* d_ws, size_t ws_size,
                              hipStream_t stream) {
    const float* x      = (const float*)d_in[0];              // [N, 128]
    const int*   eidx   = (const int*)d_in[1];                // [2, E] flat: rows then cols
    const float* eattr  = (const float*)d_in[2];              // [E]
    const int*   batch  = (const int*)d_in[3];                // [N]
    const float* W_conv = (const float*)d_in[4];              // [128, 128]
    const float* b_conv = (const float*)d_in[5];              // [128]
    const float* W_lin  = (const float*)d_in[6];              // [8, 128]
    const float* b_lin  = (const float*)d_in[7];              // [8]
    float*       outp   = (float*)d_out;                      // [64, 8]

    const int* erow = eidx;
    const int* ecol = eidx + N_EDGES;

    // ---- workspace layout ----
    char* ws = (char*)d_ws;
    const size_t SZ_H   = (size_t)N_NODES * 128 * sizeof(float);   // 51.2 MB
    float*    bufA  = (float*)(ws);
    float*    bufB  = (float*)(ws + SZ_H);
    float*    norm  = (float*)(ws + 2 * SZ_H);
    float*    dinv  = (float*)(ws + 2 * SZ_H + (size_t)N_EDGES * 4);
    _Float16* Wf16  = (_Float16*)(ws + 2 * SZ_H + (size_t)N_EDGES * 4 + (size_t)N_NODES * 4);
    float*    sums  = (float*)((char*)Wf16 + (size_t)HIDDEN * N_F * 2);
    float*    cnt   = sums + (size_t)N_GRAPHS * HIDDEN;
    _Float16* Af16  = (_Float16*)bufB;   // reused after propagation
    float*    hlin  = bufA;              // reused after f16 conversion

    const int B = 256;

    // 1) GCN norm: deg (init 1.0 self-loop) += w at col; dinv = rsqrt; per-edge norm
    k_init_deg   <<<cdiv(N_NODES, B), B, 0, stream>>>(dinv, N_NODES);
    k_deg_scatter<<<cdiv(N_EDGES, B), B, 0, stream>>>(ecol, eattr, dinv, N_EDGES);
    k_dinv       <<<cdiv(N_NODES, B), B, 0, stream>>>(dinv, N_NODES);
    k_norm       <<<cdiv(N_EDGES, B), B, 0, stream>>>(erow, ecol, eattr, dinv, norm, N_EDGES);

    // 2) K=3 propagations, ping-pong buffers (self-loop term folded into init)
    const long long n32 = (long long)N_NODES * 32;
    const long long e32 = (long long)N_EDGES * 32;
    const float* src = x;
    float* dst = bufA;
    for (int hop = 0; hop < K_HOPS; ++hop) {
        k_selfloop_init<<<cdiv(n32, B), B, 0, stream>>>(src, dinv, dst, (int)n32);
        k_edge_scatter <<<cdiv(e32, B), B, 0, stream>>>(erow, ecol, norm, src, dst, e32);
        src = dst;
        dst = (dst == bufA) ? bufB : bufA;
    }
    // final h3 is in bufA (K_HOPS odd)

    // 3) f16 conversions for the WMMA GEMM
    k_f32_to_f16<<<cdiv((long long)N_NODES * 128, B), B, 0, stream>>>(bufA, Af16, N_NODES * 128);
    k_f32_to_f16<<<cdiv(HIDDEN * N_F, B), B, 0, stream>>>(W_conv, Wf16, HIDDEN * N_F);

    // 4) h @ W_conv^T + b_conv  via v_wmma_f32_16x16x32_f16 (6250 blocks x 8 waves)
    k_wmma_gemm<<<N_NODES / 16, 256, 0, stream>>>(Af16, Wf16, b_conv, hlin);

    // 5) global mean pool
    k_zero<<<cdiv(N_GRAPHS * (HIDDEN + 1), B), B, 0, stream>>>(sums, N_GRAPHS * (HIDDEN + 1));
    k_pool<<<cdiv((long long)N_NODES * 128, B), B, 0, stream>>>(hlin, batch, sums, cnt, N_NODES * 128);

    // 6) final tiny linear -> [64, 8]
    k_final<<<cdiv(N_GRAPHS * N_PRED, B), B, 0, stream>>>(sums, cnt, W_lin, b_lin, outp);
}
